// LSTMTagger_14671608283377
// MI455X (gfx1250) — compile-verified, matrix operand order checked by reading, not run on hardware
//
#include <hip/hip_runtime.h>

// ---------------------------------------------------------------------------
// LSTM tagger for MI455X (gfx1250, wave32).
//   1) gather embedding rows -> bf16; convert W_ih/W_hh/W_out -> bf16 once
//   2) xg = x @ W_ih^T + (b_ih+b_hh)       : WMMA bf16 GEMM, 32x32/wave
//   3) sequential LSTM recurrence          : 1 WGP, K-slice of W_hh pinned in
//                                            LDS via global_load_async_to_lds
//   4) logits = hs @ W_out^T + b_out       : same WMMA GEMM (bf16 B)
//   5) in-place row-wise log_softmax
// ---------------------------------------------------------------------------

#define VOCAB 50257
#define EMB   512
#define HID   512
#define SEQT  4096
#define G4H   2048   // 4*HID
#define KPIN  64     // K-slice of W_hh pinned in LDS
#define LPAD  72     // padded LDS row stride (elements): 144B, 16B-aligned

typedef __bf16 bf16;
typedef __attribute__((ext_vector_type(16))) __bf16 v16bf;
typedef __attribute__((ext_vector_type(8)))  float  v8f;

struct alignas(16) bf8 { bf16 x[8]; };   // 16-byte vector chunk of 8 bf16

union FragBF {
    v16bf v;
    bf16  e[16];
    bf8   q[2];
};

__device__ __forceinline__ bf16 f2bf(float f) {
    unsigned int u = __builtin_bit_cast(unsigned int, f);
    unsigned int r = (u + 0x7FFFu + ((u >> 16) & 1u)) >> 16;   // RNE
    unsigned short s = (unsigned short)r;
    return __builtin_bit_cast(bf16, s);
}

__device__ __forceinline__ float bf2f(bf16 b) {
    unsigned short s = __builtin_bit_cast(unsigned short, b);
    unsigned int u = ((unsigned int)s) << 16;
    return __builtin_bit_cast(float, u);
}

__device__ __forceinline__ float sigmoidf_(float x) {
    return 1.0f / (1.0f + __expf(-x));
}

// ---------------------------------------------------------------------------
// 1) Embedding gather + fp32 -> bf16 ; bulk converts
// ---------------------------------------------------------------------------
__global__ void k_gather_embed(const int* __restrict__ sentence,
                               const float* __restrict__ emb,
                               bf16* __restrict__ xbf) {
    int idx = blockIdx.x * blockDim.x + threadIdx.x;   // [0, SEQT*EMB)
    int t = idx >> 9;          // /512
    int e = idx & 511;
    int tok = sentence[t];
    xbf[idx] = f2bf(emb[(size_t)tok * EMB + e]);
}

__global__ void k_cvt_bf16(const float* __restrict__ src,
                           bf16* __restrict__ dst, int n) {
    int i = blockIdx.x * blockDim.x + threadIdx.x;
    if (i < n) dst[i] = f2bf(src[i]);
}

// ---------------------------------------------------------------------------
// WMMA GEMM: C[M,N] = A[M,K] @ B[N,K]^T + bias1[N] + bias2[N]
// A, B bf16 row-major. Each wave computes a 32x32 super-tile (2x2 WMMA tiles,
// fragments shared across the pair -> 2 b128 loads per v_wmma). 8 waves/block
// arranged 2 (M) x 4 (N): block covers 64 x 128. K, M multiples of 32.
// ---------------------------------------------------------------------------
__global__ __launch_bounds__(256)
void k_gemm_wmma(const bf16* __restrict__ A, const bf16* __restrict__ B,
                 const float* __restrict__ bias1, const float* __restrict__ bias2,
                 float* __restrict__ C, int M, int N, int K) {
    const int lane  = threadIdx.x & 31;
    const int wid   = threadIdx.x >> 5;                 // 0..7
    const int mBase = (blockIdx.y * 2 + (wid >> 2)) * 32;
    const int nBase = (blockIdx.x * 4 + (wid & 3)) * 32;
    if (mBase >= M) return;                              // wave-uniform

    const int l16   = lane & 15;
    const int khalf = lane >> 4;                         // 0 or 1

    const int ar0 = mBase + l16;
    const int ar1 = ar0 + 16;
    int br0 = nBase + l16;      if (br0 >= N) br0 = N - 1;   // clamp pad tiles
    int br1 = nBase + 16 + l16; if (br1 >= N) br1 = N - 1;

    const bf16* a0p = A + (size_t)ar0 * K + khalf * 8;
    const bf16* a1p = A + (size_t)ar1 * K + khalf * 8;
    const bf16* b0p = B + (size_t)br0 * K + khalf * 8;
    const bf16* b1p = B + (size_t)br1 * K + khalf * 8;

    v8f acc00 = {}, acc01 = {}, acc10 = {}, acc11 = {};

    for (int k0 = 0; k0 < K; k0 += 32) {
        FragBF a0, a1, b0, b1;
        a0.q[0] = *(const bf8*)(a0p + k0);
        a0.q[1] = *(const bf8*)(a0p + k0 + 16);
        a1.q[0] = *(const bf8*)(a1p + k0);
        a1.q[1] = *(const bf8*)(a1p + k0 + 16);
        b0.q[0] = *(const bf8*)(b0p + k0);
        b0.q[1] = *(const bf8*)(b0p + k0 + 16);
        b1.q[0] = *(const bf8*)(b1p + k0);
        b1.q[1] = *(const bf8*)(b1p + k0 + 16);
        __builtin_prefetch(a0p + k0 + 64, 0, 1);         // global_prefetch_b8
        __builtin_prefetch(b0p + k0 + 64, 0, 1);
        __builtin_prefetch(b1p + k0 + 64, 0, 1);

        acc00 = __builtin_amdgcn_wmma_f32_16x16x32_bf16(
                    false, a0.v, false, b0.v, (short)0, acc00, false, false);
        acc01 = __builtin_amdgcn_wmma_f32_16x16x32_bf16(
                    false, a0.v, false, b1.v, (short)0, acc01, false, false);
        acc10 = __builtin_amdgcn_wmma_f32_16x16x32_bf16(
                    false, a1.v, false, b0.v, (short)0, acc10, false, false);
        acc11 = __builtin_amdgcn_wmma_f32_16x16x32_bf16(
                    false, a1.v, false, b1.v, (short)0, acc11, false, false);
    }

    // C/D layout: lane holds N = nB+l16, rows m = mB + 8*khalf + r
    const int mA = mBase + khalf * 8;          // tile row block for acc0x
    const int mB2 = mA + 16;                   // for acc1x
    const int n0 = nBase + l16;
    const int n1 = nBase + 16 + l16;
    if (n0 < N) {
        float bs = (bias1 ? bias1[n0] : 0.0f) + (bias2 ? bias2[n0] : 0.0f);
        #pragma unroll
        for (int r = 0; r < 8; ++r) {
            C[(size_t)(mA  + r) * N + n0] = acc00[r] + bs;
            C[(size_t)(mB2 + r) * N + n0] = acc10[r] + bs;
        }
    }
    if (n1 < N) {
        float bs = (bias1 ? bias1[n1] : 0.0f) + (bias2 ? bias2[n1] : 0.0f);
        #pragma unroll
        for (int r = 0; r < 8; ++r) {
            C[(size_t)(mA  + r) * N + n1] = acc01[r] + bs;
            C[(size_t)(mB2 + r) * N + n1] = acc11[r] + bs;
        }
    }
}

// ---------------------------------------------------------------------------
// 3) Sequential LSTM recurrence. Single workgroup (32 wave32s on one WGP).
//    h, c, gates in LDS. The K in [0,KPIN) slice of W_hh (all 2048 rows,
//    padded to LPAD-element stride) is pinned in LDS once at kernel start via
//    global_load_async_to_lds_b128 (ASYNCcnt); the K in [KPIN,512) remainder
//    streams from L2 each step. h[k] LDS reads are same-address broadcasts.
// ---------------------------------------------------------------------------
__global__ __launch_bounds__(1024)
void k_lstm_seq(const float* __restrict__ xg,        // [SEQT, 4H]
                const bf16*  __restrict__ whh,       // [4H, HID] bf16
                bf16* __restrict__ hsbf) {           // [SEQT, HID] out
    __shared__ bf16  whh_lds[G4H * LPAD];            // 288 KB pinned slice
    __shared__ float h_s[HID];
    __shared__ float c_s[HID];
    __shared__ float g_s[G4H];

    const int tid = threadIdx.x;                     // 0..1023
    if (tid < HID) { h_s[tid] = 0.0f; c_s[tid] = 0.0f; }

    // ---- async fill of the pinned W_hh K-slice: 2048 rows x 8 b128 chunks
    {
        const unsigned ldsBase = (unsigned)(size_t)(void*)&whh_lds[0];
        for (int i = tid; i < G4H * (KPIN / 8); i += 1024) {
            const int r = i >> 3;                    // row
            const int c = i & 7;                     // 16B chunk within slice
            unsigned lds = ldsBase + (unsigned)(r * (LPAD * 2) + c * 16);
            unsigned long long ga =
                (unsigned long long)(const void*)(whh + (size_t)r * HID + c * 8);
            asm volatile("global_load_async_to_lds_b128 %0, %1, off"
                         :: "v"(lds), "v"(ga) : "memory");
        }
        asm volatile("s_wait_asynccnt 0x0" ::: "memory");
    }
    __syncthreads();

    const int r0 = tid;              // gate rows owned by this thread
    const int r1 = tid + 1024;
    const bf16* l0 = whh_lds + r0 * LPAD;            // LDS-pinned K slice
    const bf16* l1 = whh_lds + r1 * LPAD;
    const bf16* w0 = whh + (size_t)r0 * HID;         // global remainder
    const bf16* w1 = whh + (size_t)r1 * HID;

    for (int t = 0; t < SEQT; ++t) {
        const float* xgt = xg + (size_t)t * G4H;
        float acc0 = xgt[r0];
        float acc1 = xgt[r1];

        // K in [0, KPIN): LDS-resident weights (ds_load_b128)
        #pragma unroll
        for (int k = 0; k < KPIN; k += 8) {
            bf8 a = *(const bf8*)(l0 + k);
            bf8 b = *(const bf8*)(l1 + k);
            #pragma unroll
            for (int u = 0; u < 8; ++u) {
                float hk = h_s[k + u];               // LDS broadcast
                acc0 = __builtin_fmaf(bf2f(a.x[u]), hk, acc0);
                acc1 = __builtin_fmaf(bf2f(b.x[u]), hk, acc1);
            }
        }
        // K in [KPIN, HID): streamed from L2
        for (int k = KPIN; k < HID; k += 8) {
            bf8 a = *(const bf8*)(w0 + k);
            bf8 b = *(const bf8*)(w1 + k);
            #pragma unroll
            for (int u = 0; u < 8; ++u) {
                float hk = h_s[k + u];
                acc0 = __builtin_fmaf(bf2f(a.x[u]), hk, acc0);
                acc1 = __builtin_fmaf(bf2f(b.x[u]), hk, acc1);
            }
        }
        g_s[r0] = acc0;
        g_s[r1] = acc1;
        __syncthreads();

        if (tid < HID) {
            float ig = sigmoidf_(g_s[tid]);
            float fg = sigmoidf_(g_s[tid + HID]);
            float gg = tanhf(g_s[tid + 2 * HID]);
            float og = sigmoidf_(g_s[tid + 3 * HID]);
            float c  = fg * c_s[tid] + ig * gg;
            c_s[tid] = c;
            float h  = og * tanhf(c);
            h_s[tid] = h;
            hsbf[(size_t)t * HID + tid] = f2bf(h);
        }
        __syncthreads();
    }
}

// ---------------------------------------------------------------------------
// 5) In-place row-wise log_softmax over VOCAB columns.
// ---------------------------------------------------------------------------
__global__ __launch_bounds__(256)
void k_log_softmax(float* __restrict__ out) {
    const int t = blockIdx.x;
    float* row = out + (size_t)t * VOCAB;
    __shared__ float red[256];
    const int tid = threadIdx.x;

    float m = -3.4e38f;
    for (int i = tid; i < VOCAB; i += 256) m = fmaxf(m, row[i]);
    red[tid] = m; __syncthreads();
    for (int s = 128; s > 0; s >>= 1) {
        if (tid < s) red[tid] = fmaxf(red[tid], red[tid + s]);
        __syncthreads();
    }
    m = red[0]; __syncthreads();

    float sum = 0.0f;
    for (int i = tid; i < VOCAB; i += 256) sum += __expf(row[i] - m);
    red[tid] = sum; __syncthreads();
    for (int s = 128; s > 0; s >>= 1) {
        if (tid < s) red[tid] += red[tid + s];
        __syncthreads();
    }
    const float lse = m + __logf(red[0]);
    __syncthreads();

    for (int i = tid; i < VOCAB; i += 256) row[i] -= lse;
}

// ---------------------------------------------------------------------------
// Host-side orchestration
// ---------------------------------------------------------------------------
extern "C" void kernel_launch(void* const* d_in, const int* in_sizes, int n_in,
                              void* d_out, int out_size, void* d_ws, size_t ws_size,
                              hipStream_t stream) {
    (void)in_sizes; (void)n_in; (void)out_size; (void)ws_size;

    const int*   sentence = (const int*)  d_in[0];
    const float* emb      = (const float*)d_in[1];
    const float* W_ih     = (const float*)d_in[2];
    const float* W_hh     = (const float*)d_in[3];
    const float* b_ih     = (const float*)d_in[4];
    const float* b_hh     = (const float*)d_in[5];
    const float* W_out    = (const float*)d_in[6];
    const float* b_out    = (const float*)d_in[7];
    float*       out      = (float*)d_out;

    // Workspace carve-up (all 256B aligned by construction)
    char* ws = (char*)d_ws;
    bf16*  xbf    = (bf16*)ws;  ws += (size_t)SEQT  * EMB * sizeof(bf16);  // 4 MB
    bf16*  wihbf  = (bf16*)ws;  ws += (size_t)G4H   * EMB * sizeof(bf16);  // 2 MB
    bf16*  whhbf  = (bf16*)ws;  ws += (size_t)G4H   * HID * sizeof(bf16);  // 2 MB
    bf16*  hsbf   = (bf16*)ws;  ws += (size_t)SEQT  * HID * sizeof(bf16);  // 4 MB
    bf16*  woutbf = (bf16*)ws;  ws += ((size_t)VOCAB * HID * sizeof(bf16) + 255) & ~(size_t)255; // 51.5 MB
    float* xg     = (float*)ws; ws += (size_t)SEQT  * G4H * sizeof(float); // 33.5 MB

    // 1) gather + one-time bf16 conversions (W_out bf16 stays L2-resident)
    k_gather_embed<<<(SEQT * EMB) / 256, 256, 0, stream>>>(sentence, emb, xbf);
    k_cvt_bf16<<<(G4H * EMB + 255) / 256, 256, 0, stream>>>(W_ih, wihbf, G4H * EMB);
    k_cvt_bf16<<<(G4H * HID + 255) / 256, 256, 0, stream>>>(W_hh, whhbf, G4H * HID);
    k_cvt_bf16<<<(VOCAB * HID + 255) / 256, 256, 0, stream>>>(W_out, woutbf, VOCAB * HID);

    // 2) xg = x @ W_ih^T + (b_ih + b_hh)   [4096 x 2048]
    {
        dim3 grid(G4H / 128, SEQT / 64);
        k_gemm_wmma<<<grid, 256, 0, stream>>>(xbf, wihbf, b_ih, b_hh,
                                              xg, SEQT, G4H, EMB);
    }

    // 3) sequential recurrence -> hs (bf16)
    k_lstm_seq<<<1, 1024, 0, stream>>>(xg, whhbf, hsbf);

    // 4) logits = hs @ W_out^T + b_out   [4096 x 50257]
    {
        dim3 grid((VOCAB + 127) / 128, SEQT / 64);
        k_gemm_wmma<<<grid, 256, 0, stream>>>(hsbf, woutbf, b_out,
                                              (const float*)nullptr,
                                              out, SEQT, VOCAB, HID);
    }

    // 5) log_softmax in place
    k_log_softmax<<<SEQT, 256, 0, stream>>>(out);
}
